// Transducer_3530463118076
// MI455X (gfx1250) — matrix operand order repaired and checked by reading, not compiled, not run on hardware
//
#include <hip/hip_runtime.h>
#include <hip/hip_bf16.h>
#include <math.h>

typedef __bf16 bf16;
typedef __attribute__((ext_vector_type(16))) __bf16 v16bf;
typedef __attribute__((ext_vector_type(8)))  float  v8f;

__device__ __forceinline__ float sigf(float x) { return 1.0f / (1.0f + __expf(-x)); }

// ---- CDNA5 async global->LDS staging (ASYNCcnt-tracked DMA, no VGPR roundtrip)
#if defined(__gfx1250__) && __has_builtin(__builtin_amdgcn_global_load_async_to_lds_b128)
#define HAVE_ASYNC_LDS 1
#else
#define HAVE_ASYNC_LDS 0
#endif

#if HAVE_ASYNC_LDS
typedef int v4i __attribute__((__vector_size__(16)));
typedef __attribute__((address_space(1))) v4i* as1_v4i_ptr;
typedef __attribute__((address_space(3))) v4i* as3_v4i_ptr;
#endif

__device__ __forceinline__ void stage16(const bf16* g, bf16* l) {
#if HAVE_ASYNC_LDS
    __builtin_amdgcn_global_load_async_to_lds_b128(
        (as1_v4i_ptr)(void*)g, (as3_v4i_ptr)(void*)l, 0, 0);
#else
    *(float4*)l = *(const float4*)g;
#endif
}

__device__ __forceinline__ void wait_async() {
#if HAVE_ASYNC_LDS && __has_builtin(__builtin_amdgcn_s_wait_asynccnt)
    __builtin_amdgcn_s_wait_asynccnt(0);
#endif
}

// ---------------------------------------------------------------------------
// bf16 GEMM: C[M,N](f32) = A[M,K](bf16) * Bt[N,K](bf16)^T (+bias) (+C if acc)
// B is pre-transposed [N,K] so BOTH tiles stage as contiguous 16B vectors via
// global_load_async_to_lds_b128 into double-buffered LDS (one barrier/K-step).
// Out-of-range rows/cols are clamped during staging (their products only land
// in C rows/cols the store masks off). K must be a multiple of 32.
// 128 threads = 4 waves:
//   <64,128> : each wave owns a 16x128 strip -> 8 WMMA / K-step   (big GEMMs)
//   <16,256> : waves split N, 16x64 strip each -> 4 WMMA / K-step (M=4 steps)
// ---------------------------------------------------------------------------
template <int BM, int BN>
__global__ __launch_bounds__(128)
void k_gemm_bf16(const bf16* __restrict__ A, int lda,
                 const bf16* __restrict__ Bt, int ldb,
                 float* __restrict__ C, int ldc,
                 const float* __restrict__ bias,
                 int M, int N, int K, int acc_flag)
{
    constexpr int WR  = BM / 16;            // wave rows (1 or 4)
    constexpr int WCW = 4 / WR;             // waves along cols
    constexpr int CT  = BN / (16 * WCW);    // 16-col tiles per wave
    constexpr int nAv = BM * 4;             // 16B vectors in A tile (BM x 32)
    constexpr int nBv = BN * 4;
    constexpr int NV  = nAv + nBv;

    __shared__ __align__(16) bf16 As[2][BM][32];
    __shared__ __align__(16) bf16 Bs[2][BN][32];

    const int tid   = threadIdx.x;
    const int wave  = tid >> 5;
    const int lane  = tid & 31;
    const int lhalf = lane >> 4;
    const int l15   = lane & 15;
    const int bm    = blockIdx.y * BM;
    const int bn    = blockIdx.x * BN;
    const int wrow  = (wave % WR) * 16;
    const int wcol  = (wave / WR) * (CT * 16);

    auto stage_tile = [&](int sb, int k0) {
#pragma unroll
        for (int v = tid; v < NV; v += 128) {
            if (v < nAv) {
                int r = v >> 2, seg = v & 3;
                int gr = bm + r; gr = (gr < M) ? gr : (M - 1);      // clamp
                stage16(A + (size_t)gr * lda + k0 + seg * 8, &As[sb][r][seg * 8]);
            } else {
                int v2 = v - nAv;
                int n = v2 >> 2, seg = v2 & 3;
                int gn = bn + n; gn = (gn < N) ? gn : (N - 1);      // clamp
                stage16(Bt + (size_t)gn * ldb + k0 + seg * 8, &Bs[sb][n][seg * 8]);
            }
        }
    };

    v8f acc[CT];
#pragma unroll
    for (int i = 0; i < CT; ++i) acc[i] = (v8f){0.f,0.f,0.f,0.f,0.f,0.f,0.f,0.f};

    int buf = 0;
    stage_tile(0, 0);
    wait_async();
    __syncthreads();

    for (int k0 = 0; k0 < K; k0 += 32) {
        if (k0 + 32 < K) stage_tile(buf ^ 1, k0 + 32);   // async DMA overlaps WMMA burst

        // A fragment: lanes0-15 K 0..7 & 16..23, lanes16-31 K 8..15 & 24..31
        const bf16* ap = &As[buf][wrow + l15][0];
        union { float4 f[2]; v16bf v; } ua;
        ua.f[0] = *(const float4*)(ap + lhalf * 8);
        ua.f[1] = *(const float4*)(ap + lhalf * 8 + 16);
#pragma unroll
        for (int nt = 0; nt < CT; ++nt) {
            // B fragment: col = l15 in tile; lanes0-15 K 0..7 & 16..23, lanes16-31 K 8..15 & 24..31
            const bf16* bp = &Bs[buf][wcol + nt * 16 + l15][0];
            union { float4 f[2]; v16bf v; } ub;
            ub.f[0] = *(const float4*)(bp + lhalf * 8);
            ub.f[1] = *(const float4*)(bp + lhalf * 8 + 16);
            acc[nt] = __builtin_amdgcn_wmma_f32_16x16x32_bf16(
                false, ua.v, false, ub.v, (short)0, acc[nt], false, false);
        }

        wait_async();       // own DMA for next tile landed
        __syncthreads();    // all waves: next buffer visible, current buffer free
        buf ^= 1;
    }

    const int row0 = bm + wrow + lhalf * 8;
#pragma unroll
    for (int nt = 0; nt < CT; ++nt) {
        int gc = bn + wcol + nt * 16 + l15;
        if (gc >= N) continue;
        float bv = bias ? bias[gc] : 0.0f;
#pragma unroll
        for (int r = 0; r < 8; ++r) {
            int gr = row0 + r;
            if (gr < M) {
                size_t off = (size_t)gr * (size_t)ldc + (size_t)gc;
                float v = acc[nt][r] + bv;
                if (acc_flag) v += C[off];
                C[off] = v;
            }
        }
    }
}

// ---------------------------------------------------------------------------
// Elementwise / support kernels
// ---------------------------------------------------------------------------
__global__ void k_cvt_bf16(bf16* __restrict__ dst, const float* __restrict__ src, int n) {
    int i = blockIdx.x * blockDim.x + threadIdx.x;
    if (i < n) dst[i] = (bf16)src[i];
}

// transpose+convert weight: src[K,N] f32 -> dst[N,Kpad] bf16 (zero-pad K..Kpad)
__global__ void k_cvt_transpose(bf16* __restrict__ dst, const float* __restrict__ src,
                                int N, int K, int Kpad) {
    int i = blockIdx.x * blockDim.x + threadIdx.x;
    if (i >= N * Kpad) return;
    int n = i / Kpad, k = i % Kpad;
    dst[i] = (k < K) ? (bf16)src[(size_t)k * N + n] : (bf16)0.0f;
}

// convert with column zero-padding: src[rows,cols] -> dst[rows,colspad]
__global__ void k_cvt_pad(bf16* __restrict__ dst, const float* __restrict__ src,
                          int rows, int cols, int colspad) {
    int i = blockIdx.x * blockDim.x + threadIdx.x;
    if (i >= rows * colspad) return;
    int r = i / colspad, c = i % colspad;
    dst[i] = (c < cols) ? (bf16)src[(size_t)r * cols + c] : (bf16)0.0f;
}

__global__ void k_zero_bf16(bf16* __restrict__ dst, int n) {
    int i = blockIdx.x * blockDim.x + threadIdx.x;
    if (i < n) dst[i] = (bf16)0.0f;
}

__global__ void k_zero_f32(float* __restrict__ dst, int n) {
    int i = blockIdx.x * blockDim.x + threadIdx.x;
    if (i < n) dst[i] = 0.0f;
}

__global__ void k_copy_f32(float* __restrict__ dst, const float* __restrict__ src, int n) {
    int i = blockIdx.x * blockDim.x + threadIdx.x;
    if (i < n) dst[i] = src[i];
}

// 4 strided rows -> packed bf16 (rows 4..15 of dst pre-zeroed elsewhere)
__global__ void k_rows4_bf16(bf16* __restrict__ dst, const float* __restrict__ src,
                             int cols, int src_ld) {
    int i = blockIdx.x * blockDim.x + threadIdx.x;
    if (i >= 4 * cols) return;
    int r = i / cols, c = i % cols;
    dst[r * cols + c] = (bf16)src[(size_t)r * src_ld + c];
}

__global__ void k_rows4_f32(float* __restrict__ dst, const float* __restrict__ src,
                            int cols, int src_ld) {
    int i = blockIdx.x * blockDim.x + threadIdx.x;
    if (i >= 4 * cols) return;
    int r = i / cols, c = i % cols;
    dst[r * cols + c] = src[(size_t)r * src_ld + c];
}

// LSTMP gates: tf.v1 order i,j,f,o; forget bias +1. z row stride zs per batch.
__global__ void k_lstmp_gates(const float* __restrict__ z, int zs,
                              float* __restrict__ c, bf16* __restrict__ mpad) {
    int i = blockIdx.x * blockDim.x + threadIdx.x;
    if (i >= 4 * 2048) return;
    int b = i >> 11, u = i & 2047;
    const float* zb = z + (size_t)b * zs;
    float gi = zb[u], gj = zb[2048 + u], gf = zb[4096 + u], go = zb[6144 + u];
    float cc = sigf(gf + 1.0f) * c[i] + sigf(gi) * tanhf(gj);
    c[i] = cc;
    mpad[i] = (bf16)(sigf(go) * tanhf(cc));
}

// keras LSTM gates: order i,f,g,o; no forget bias.
__global__ void k_lstm_gates(const float* __restrict__ z, int zs,
                             float* __restrict__ c,
                             float* __restrict__ hout, int hstride,
                             bf16* __restrict__ hpad) {
    int i = blockIdx.x * blockDim.x + threadIdx.x;
    if (i >= 4 * 1024) return;
    int b = i >> 10, u = i & 1023;
    const float* zb = z + (size_t)b * zs;
    float gi = zb[u], gf = zb[1024 + u], gg = zb[2048 + u], go = zb[3072 + u];
    float cc = sigf(gf) * c[i] + sigf(gi) * tanhf(gg);
    c[i] = cc;
    float h = sigf(go) * tanhf(cc);
    hout[(size_t)b * hstride + u] = h;
    hpad[i] = (bf16)h;
}

__global__ __launch_bounds__(256)
void k_layernorm(float* __restrict__ x, const float* __restrict__ g,
                 const float* __restrict__ b, int cols) {
    float* xr = x + (size_t)blockIdx.x * cols;
    __shared__ float red[256];
    int t = threadIdx.x;
    float s = 0.f;
    for (int c = t; c < cols; c += 256) s += xr[c];
    red[t] = s; __syncthreads();
    for (int o = 128; o > 0; o >>= 1) { if (t < o) red[t] += red[t + o]; __syncthreads(); }
    float mean = red[0] / (float)cols;
    __syncthreads();
    float v = 0.f;
    for (int c = t; c < cols; c += 256) { float d = xr[c] - mean; v += d * d; }
    red[t] = v; __syncthreads();
    for (int o = 128; o > 0; o >>= 1) { if (t < o) red[t] += red[t + o]; __syncthreads(); }
    float inv = rsqrtf(red[0] / (float)cols + 1e-3f);
    __syncthreads();
    for (int c = t; c < cols; c += 256) xr[c] = (xr[c] - mean) * inv * g[c] + b[c];
}

__global__ __launch_bounds__(256)
void k_softmax_bf16(float* __restrict__ s, bf16* __restrict__ out, int cols) {
    float* xr = s + (size_t)blockIdx.x * cols;
    bf16* orow = out + (size_t)blockIdx.x * cols;
    __shared__ float red[256];
    int t = threadIdx.x;
    float mx = -3.0e38f;
    for (int c = t; c < cols; c += 256) mx = fmaxf(mx, xr[c]);
    red[t] = mx; __syncthreads();
    for (int o = 128; o > 0; o >>= 1) { if (t < o) red[t] = fmaxf(red[t], red[t + o]); __syncthreads(); }
    mx = red[0]; __syncthreads();
    float sum = 0.f;
    for (int c = t; c < cols; c += 256) { float e = __expf(xr[c] - mx); xr[c] = e; sum += e; }
    red[t] = sum; __syncthreads();
    for (int o = 128; o > 0; o >>= 1) { if (t < o) red[t] += red[t + o]; __syncthreads(); }
    float inv = 1.0f / red[0];
    __syncthreads();
    for (int c = t; c < cols; c += 256) orow[c] = (bf16)(xr[c] * inv);
}

__global__ void k_embed_bf16(bf16* __restrict__ dst, const int* __restrict__ ids,
                             const float* __restrict__ emb, int n) {
    int i = blockIdx.x * blockDim.x + threadIdx.x;
    if (i >= n) return;
    int r = i >> 6, e = i & 63;
    dst[i] = (bf16)emb[ids[r] * 64 + e];
}

// joint rows: dst[r][k] = enc[b,t,k] + pred[b,u,k], r_global = b*(127*64)+t*64+u
__global__ void k_joint_rows(bf16* __restrict__ dst, const float* __restrict__ enc,
                             const float* __restrict__ pred, int base, int total) {
    int i = blockIdx.x * blockDim.x + threadIdx.x;
    if (i >= 1024 * 1024) return;
    int r = i >> 10, k = i & 1023;
    int gr = base + r;
    float v = 0.f;
    if (gr < total) {
        int b = gr / (127 * 64);
        int rem = gr % (127 * 64);
        int tt = rem >> 6, u = rem & 63;
        v = enc[((size_t)b * 127 + tt) * 1024 + k] + pred[((size_t)b * 64 + u) * 1024 + k];
    }
    dst[i] = (bf16)v;
}

// ---------------------------------------------------------------------------
// Host orchestration
// ---------------------------------------------------------------------------
static inline size_t alignup(size_t x) { return (x + 255) & ~(size_t)255; }

extern "C" void kernel_launch(void* const* d_in, const int* in_sizes, int n_in,
                              void* d_out_v, int out_size, void* d_ws, size_t ws_size,
                              hipStream_t stream)
{
    (void)in_sizes; (void)n_in; (void)out_size; (void)ws_size;
    float* d_out = (float*)d_out_v;

    // ---- input pointers, setup_inputs() insertion order
    const float* enc_inp   = (const float*)d_in[0];  // [4,128,80]
    const int*   pred_inp  = (const int*)d_in[1];    // [4,64]
    const float* enc_state = (const float*)d_in[2];  // [2,4,2048]
    int idx = 3;
    const float *eWx[8], *eWh[8], *eb[8], *eWp[8];
    for (int i = 0; i < 8; ++i) {
        eWx[i] = (const float*)d_in[idx++];
        eWh[i] = (const float*)d_in[idx++];
        eb[i]  = (const float*)d_in[idx++];
        eWp[i] = (const float*)d_in[idx++];
    }
    const float* ln_g   = (const float*)d_in[idx++];
    const float* ln_b   = (const float*)d_in[idx++];
    const float* conv_k = (const float*)d_in[idx++]; // [2,1024,1024]
    const float* conv_b = (const float*)d_in[idx++];
    const float* embedw = (const float*)d_in[idx++]; // [29,64]
    const float *pWx[2], *pWh[2], *pb[2];
    for (int i = 0; i < 2; ++i) {
        pWx[i] = (const float*)d_in[idx++];
        pWh[i] = (const float*)d_in[idx++];
        pb[i]  = (const float*)d_in[idx++];
    }
    const float* pln_g = (const float*)d_in[idx++];
    const float* pln_b = (const float*)d_in[idx++];
    const float* Wj    = (const float*)d_in[idx++];
    const float* bj    = (const float*)d_in[idx++];
    const float* Wsm   = (const float*)d_in[idx++];
    const float* bsm   = (const float*)d_in[idx++];
    const float* Wproj = (const float*)d_in[idx++];
    const float* bproj = (const float*)d_in[idx++];

    // ---- workspace carve-out (~100 MB)
    char* w = (char*)d_ws;
    size_t off = 0;
    auto carve = [&](size_t bytes) { void* p = w + off; off = alignup(off + bytes); return p; };
    bf16*  Wxb    = (bf16*)carve((size_t)8192 * 1024 * 2);   // [N=8192, Kpad] transposed
    bf16*  Whb    = (bf16*)carve((size_t)8192 * 1024 * 2);   // [8192, 1024]
    bf16*  Wpb    = (bf16*)carve((size_t)1024 * 2048 * 2);   // [1024, 2048]; also conv [2x 1024,1024]
    float* xW     = (float*)carve((size_t)4 * 128 * 8192 * 4);
    float* XA     = (float*)carve((size_t)4 * 128 * 1024 * 4);
    float* XB     = (float*)carve((size_t)4 * 128 * 1024 * 4);
    bf16*  xb     = (bf16*)carve((size_t)4 * 128 * 1024 * 2);
    bf16*  hpad   = (bf16*)carve((size_t)16 * 1024 * 2);
    bf16*  mpad   = (bf16*)carve((size_t)16 * 2048 * 2);
    float* cbuf   = (float*)carve((size_t)4 * 2048 * 4);
    float* pseq   = (float*)carve((size_t)4 * 64 * 1024 * 4);
    bf16*  pbin   = (bf16*)carve((size_t)4 * 64 * 1024 * 2);
    float* cP     = (float*)carve((size_t)4 * 1024 * 4);
    bf16*  jointA = (bf16*)carve((size_t)1024 * 1024 * 2);
    float* jbuf   = (float*)carve((size_t)1024 * 640 * 4);
    bf16*  jb16   = (bf16*)carve((size_t)1024 * 640 * 2);
    float* sbuf   = (float*)carve((size_t)1024 * 4096 * 4);
    bf16*  sb16   = (bf16*)carve((size_t)1024 * 4096 * 2);
    bf16*  Wjt    = (bf16*)carve((size_t)640 * 1024 * 2);    // [640, 1024]
    bf16*  Wsmt   = (bf16*)carve((size_t)4096 * 640 * 2);    // [4096, 640]
    bf16*  Wprojt = (bf16*)carve((size_t)29 * 4096 * 2);     // [29, 4096]

    auto gemm = [&](const bf16* A, int lda, const bf16* Bt, int ldb, float* C, int ldc,
                    const float* bias, int M, int N, int K, int accf) {
        if (M <= 16) {
            dim3 g((N + 255) / 256, 1);
            k_gemm_bf16<16, 256><<<g, 128, 0, stream>>>(A, lda, Bt, ldb, C, ldc, bias, M, N, K, accf);
        } else {
            dim3 g((N + 127) / 128, (M + 63) / 64);
            k_gemm_bf16<64, 128><<<g, 128, 0, stream>>>(A, lda, Bt, ldb, C, ldc, bias, M, N, K, accf);
        }
    };
    auto cvt = [&](bf16* dst, const float* src, int n) {
        k_cvt_bf16<<<(n + 255) / 256, 256, 0, stream>>>(dst, src, n);
    };
    auto cvtT = [&](bf16* dst, const float* src, int N, int K, int Kpad) {
        int n = N * Kpad;
        k_cvt_transpose<<<(n + 255) / 256, 256, 0, stream>>>(dst, src, N, K, Kpad);
    };

    // ================= Encoder =================
    k_zero_bf16<<<(16 * 1024 + 255) / 256, 256, 0, stream>>>(hpad, 16 * 1024);
    k_zero_bf16<<<(16 * 2048 + 255) / 256, 256, 0, stream>>>(mpad, 16 * 2048);
    k_copy_f32<<<(8192 + 255) / 256, 256, 0, stream>>>(cbuf, enc_state, 8192);        // c0
    k_rows4_bf16<<<(4 * 1024 + 255) / 256, 256, 0, stream>>>(hpad, enc_state + 4 * 2048,
                                                             1024, 2048);             // h0 = state[1][:, :1024]

    const float* curf = nullptr;
    int Tcur = 128;
    for (int L = 0; L < 8; ++L) {
        int in_dim = (L == 0) ? 80 : 1024;
        int Kp     = (L == 0) ? 96 : 1024;     // K padded to multiple of 32
        int Mrows  = 4 * Tcur;
        float* outb = (curf == XA) ? XB : XA;

        cvtT(Wxb, eWx[L], 8192, in_dim, Kp);
        cvtT(Whb, eWh[L], 8192, 1024, 1024);
        cvtT(Wpb, eWp[L], 1024, 2048, 2048);

        if (L == 0) {
            k_cvt_pad<<<(Mrows * 96 + 255) / 256, 256, 0, stream>>>(xb, enc_inp, Mrows, 80, 96);
        } else {
            cvt(xb, curf, Mrows * 1024);
        }
        // xW = x @ Wx + b    [Mrows, 8192]
        gemm(xb, Kp, Wxb, Kp, xW, 8192, eb[L], Mrows, 8192, Kp, 0);

        for (int t = 0; t < Tcur; ++t) {
            float* zt = xW + (size_t)t * 8192;
            // z = xW_t + h @ Wh  (accumulate in place; batch stride Tcur*8192)
            gemm(hpad, 1024, Whb, 1024, zt, Tcur * 8192, nullptr, 4, 8192, 1024, 1);
            k_lstmp_gates<<<(4 * 2048 + 255) / 256, 256, 0, stream>>>(zt, Tcur * 8192, cbuf, mpad);
            // h2 = m @ Wp  -> x_out[:, t, :]
            float* ht = outb + (size_t)t * 1024;
            gemm(mpad, 2048, Wpb, 2048, ht, Tcur * 1024, nullptr, 4, 1024, 2048, 0);
            k_rows4_bf16<<<(4 * 1024 + 255) / 256, 256, 0, stream>>>(hpad, ht, 1024, Tcur * 1024);
        }

        if (L == 7) {
            // final hT (pre-LayerNorm) -> d_out tail
            k_rows4_f32<<<(4 * 1024 + 255) / 256, 256, 0, stream>>>(
                d_out + 942848 + 8192, outb + (size_t)(Tcur - 1) * 1024, 1024, Tcur * 1024);
        }

        k_layernorm<<<Mrows, 256, 0, stream>>>(outb, ln_g, ln_b, 1024);
        curf = outb;

        if (L == 1) {
            // Conv1D time reduction: y[:,t] = x[:,t]@K0 + x[:,t+1]@K1 + cb  (T: 128 -> 127)
            cvt(xb, curf, Mrows * 1024);
            cvtT(Wpb, conv_k, 1024, 1024, 1024);                           // K0^T
            cvtT(Wpb + (size_t)1024 * 1024, conv_k + (size_t)1024 * 1024,
                 1024, 1024, 1024);                                        // K1^T
            const bf16* K0t = Wpb;
            const bf16* K1t = Wpb + (size_t)1024 * 1024;
            float* convout = (curf == XA) ? XB : XA;
            for (int b2 = 0; b2 < 4; ++b2) {
                const bf16* a0 = xb + (size_t)b2 * Tcur * 1024;
                float* co = convout + (size_t)b2 * (Tcur - 1) * 1024;
                gemm(a0, 1024, K0t, 1024, co, 1024, conv_b, Tcur - 1, 1024, 1024, 0);
                gemm(a0 + 1024, 1024, K1t, 1024, co, 1024, nullptr, Tcur - 1, 1024, 1024, 1);
            }
            Tcur = 127;
            curf = convout;
        }
    }
    const float* enc_out = curf;  // [4,127,1024]
    // final cT -> d_out tail
    k_copy_f32<<<(8192 + 255) / 256, 256, 0, stream>>>(d_out + 942848, cbuf, 8192);

    // ================= Prediction network =================
    k_embed_bf16<<<(4 * 64 * 64 + 255) / 256, 256, 0, stream>>>(pbin, pred_inp, embedw, 4 * 64 * 64);
    for (int L = 0; L < 2; ++L) {
        int in_dim = (L == 0) ? 64 : 1024;
        cvtT(Wxb, pWx[L], 4096, in_dim, in_dim);
        cvtT(Whb, pWh[L], 4096, 1024, 1024);
        gemm(pbin, in_dim, Wxb, in_dim, xW, 4096, pb[L], 256, 4096, in_dim, 0);
        k_zero_f32<<<(4096 + 255) / 256, 256, 0, stream>>>(cP, 4096);
        k_zero_bf16<<<(16 * 1024 + 255) / 256, 256, 0, stream>>>(hpad, 16 * 1024);
        for (int t = 0; t < 64; ++t) {
            float* zt = xW + (size_t)t * 4096;
            gemm(hpad, 1024, Whb, 1024, zt, 64 * 4096, nullptr, 4, 4096, 1024, 1);
            k_lstm_gates<<<(4 * 1024 + 255) / 256, 256, 0, stream>>>(
                zt, 64 * 4096, cP, pseq + (size_t)t * 1024, 64 * 1024, hpad);
        }
        k_layernorm<<<256, 256, 0, stream>>>(pseq, pln_g, pln_b, 1024);
        if (L == 0) cvt(pbin, pseq, 256 * 1024);
    }

    // ================= Joint network (chunked) =================
    cvtT(Wjt, Wj, 640, 1024, 1024);
    cvtT(Wsmt, Wsm, 4096, 640, 640);
    cvtT(Wprojt, Wproj, 29, 4096, 4096);
    const int TOTAL = 4 * 127 * 64;  // 32512
    for (int c0 = 0; c0 < TOTAL; c0 += 1024) {
        int rows = (TOTAL - c0 < 1024) ? (TOTAL - c0) : 1024;
        k_joint_rows<<<(1024 * 1024 + 255) / 256, 256, 0, stream>>>(jointA, enc_out, pseq, c0, TOTAL);
        gemm(jointA, 1024, Wjt, 1024, jbuf, 640, bj, rows, 640, 1024, 0);
        cvt(jb16, jbuf, rows * 640);
        gemm(jb16, 640, Wsmt, 640, sbuf, 4096, bsm, rows, 4096, 640, 0);
        k_softmax_bf16<<<rows, 256, 0, stream>>>(sbuf, sb16, 4096);
        gemm(sb16, 4096, Wprojt, 4096, d_out + (size_t)c0 * 29, 29, bproj, rows, 29, 4096, 0);
    }
}
